// Attention_72722386256499
// MI455X (gfx1250) — matrix operand order repaired and checked by reading, not compiled, not run on hardware
//
#include <hip/hip_runtime.h>
#include <hip/hip_bf16.h>
#include <stdint.h>

#define HEADS 8
#define DHEAD 64
#define DIMC  512
#define HID   512
#define BATCH 4
#define SEQ   2048

typedef __attribute__((ext_vector_type(16))) __bf16 v16bf;
typedef __attribute__((ext_vector_type(8)))  float  v8f;

union FragB { v16bf v; uint4 q[2]; unsigned short u[16]; };

__device__ __forceinline__ unsigned short f2bf(float x){
  unsigned u = __float_as_uint(x);
  u += 0x7fffu + ((u >> 16) & 1u);           // round-to-nearest-even
  return (unsigned short)(u >> 16);
}

__device__ __forceinline__ v8f wmma_bf16(const FragB& a, const FragB& b, v8f c){
  return __builtin_amdgcn_wmma_f32_16x16x32_bf16(false, a.v, false, b.v,
                                                 (short)0, c, false, false);
}

// Async global->LDS copy, 16 bytes per lane, tracked by ASYNCcnt.
// Low 32 bits of a generic shared pointer are the LDS byte offset (ISA 10.2).
__device__ __forceinline__ void async_copy_b128(void* lds, const void* gaddr){
  asm volatile("global_load_async_to_lds_b128 %0, %1, off"
               :: "v"((unsigned)(uintptr_t)lds),
                  "v"((unsigned long long)(uintptr_t)gaddr)
               : "memory");
}

// ---------------- prep kernels ----------------
__global__ void cvt_f32_bf16(const float* __restrict__ src,
                             unsigned short* __restrict__ dst, int n){
  int i = blockIdx.x * blockDim.x + threadIdx.x;
  if (i < n) dst[i] = f2bf(src[i]);
}

// x: (B, DIMC, SEQ) f32 -> xT: (B, SEQ, DIMC) bf16
__global__ void transpose_x(const float* __restrict__ x,
                            unsigned short* __restrict__ xT){
  int i = blockIdx.x * blockDim.x + threadIdx.x;
  if (i >= BATCH * DIMC * SEQ) return;
  int l = i % SEQ;
  int c = (i / SEQ) % DIMC;
  int b = i / (SEQ * DIMC);
  xT[((size_t)b * SEQ + l) * DIMC + c] = f2bf(x[i]);
}

// ---------------- QKV projection ----------------
// qkv[o][l] = sum_c w[o][c] * x[c][l]; A = w_qkv (O,C) bf16, B = xT (L,C) bf16.
// One wave computes a 16(o) x 32(l) tile; writes Q (L,D)*scale, K (L,D), V (D,L).
__global__ void qkv_gemm(const unsigned short* __restrict__ wb,
                         const unsigned short* __restrict__ xT,
                         unsigned short* __restrict__ Qs,
                         unsigned short* __restrict__ Kb,
                         unsigned short* __restrict__ Vb){
  const int lane = threadIdx.x & 31;
  const int wave = threadIdx.x >> 5;
  int task = blockIdx.x * (blockDim.x >> 5) + wave;      // B * 96 * 64 tasks
  int b  = task / (96 * 64);
  int rm = task % (96 * 64);
  int ot = (rm / 64) * 16;
  int lt = (rm % 64) * 32;
  const int hi  = (lane >= 16);
  const int l15 = lane & 15;

  v8f acc0 = {}, acc1 = {};
  const int arow = ot + l15;
  for (int c = 0; c < DIMC; c += 32){
    FragB fa, fb0, fb1;
    fa.q[0] = *(const uint4*)&wb[arow * DIMC + c + (hi ? 16 : 0)];
    fa.q[1] = *(const uint4*)&wb[arow * DIMC + c + (hi ? 16 : 0) + 8];
    size_t r0 = ((size_t)b * SEQ + lt + l15) * DIMC + c + (hi ? 16 : 0);
    fb0.q[0] = *(const uint4*)&xT[r0];
    fb0.q[1] = *(const uint4*)&xT[r0 + 8];
    size_t r1 = ((size_t)b * SEQ + lt + 16 + l15) * DIMC + c + (hi ? 16 : 0);
    fb1.q[0] = *(const uint4*)&xT[r1];
    fb1.q[1] = *(const uint4*)&xT[r1 + 8];
    acc0 = wmma_bf16(fa, fb0, acc0);
    acc1 = wmma_bf16(fa, fb1, acc1);
  }

  const float scale = 0.125f;                            // 64^-0.5
  for (int t = 0; t < 2; ++t){
    int l = lt + t * 16 + l15;
    v8f acc = t ? acc1 : acc0;
    for (int r = 0; r < 8; ++r){
      int o   = ot + r + (hi ? 8 : 0);
      int sel = o / HID;
      int oh  = o % HID;
      int h = oh / DHEAD, d = oh % DHEAD;
      int bh = b * HEADS + h;
      float v = acc[r];
      if (sel == 0)      Qs[((size_t)bh * SEQ + l) * DHEAD + d] = f2bf(v * scale);
      else if (sel == 1) Kb[((size_t)bh * SEQ + l) * DHEAD + d] = f2bf(v);
      else               Vb[((size_t)bh * DHEAD + d) * SEQ + l] = f2bf(v);
    }
  }
}

// ---------------- flash attention ----------------
// Per workgroup: one (b,h), 8 waves x 16 query columns (128 queries/WG).
// K/V chunks double-buffered in LDS via global_load_async_to_lds_b128.
// Computes S^T tiles (M=j, N=i) so softmax over j is lane-local.
__global__ void flash_attn(const unsigned short* __restrict__ Qs,
                           const unsigned short* __restrict__ Kb,
                           const unsigned short* __restrict__ Vb,
                           unsigned short* __restrict__ attnT){
  alignas(16) __shared__ unsigned short sK[2][32 * 64];  // (j, d) double-buffered
  alignas(16) __shared__ unsigned short sV[2][64 * 32];  // (d, j) double-buffered
  alignas(16) __shared__ unsigned short sP[8][16 * 32];  // per-wave (i, j)

  const int lane = threadIdx.x & 31;
  const int wave = threadIdx.x >> 5;
  const int tid  = threadIdx.x;
  const int nib  = SEQ / 128;
  int bh = blockIdx.x / nib;
  int ib = blockIdx.x % nib;
  const int hi  = (lane >= 16);
  const int l15 = lane & 15;
  int i = ib * 128 + wave * 16 + l15;                    // query position

  // Q B-fragments (reused for entire j loop)
  FragB q0, q1;
  {
    const unsigned short* qp = &Qs[((size_t)bh * SEQ + i) * DHEAD];
    q0.q[0] = *(const uint4*)&qp[(hi ? 16 : 0)];
    q0.q[1] = *(const uint4*)&qp[(hi ? 16 : 0) + 8];
    q1.q[0] = *(const uint4*)&qp[32 + (hi ? 16 : 0)];
    q1.q[1] = *(const uint4*)&qp[32 + (hi ? 16 : 0) + 8];
  }

  v8f acc[4] = {{}, {}, {}, {}};
  float mrun = -3.0e38f, lrun = 0.0f;

  const unsigned short* gK = &Kb[(size_t)bh * SEQ * DHEAD];
  const unsigned short* gV = &Vb[(size_t)bh * DHEAD * SEQ];

  // per-thread cooperative copy addressing (256 threads, 1 b128 each for K and V)
  const int eK = tid * 8;                                // K chunk: contiguous 32x64
  const int dV = tid >> 2, cV = (tid & 3) * 8;           // V chunk: 64 rows x 32 cols

  auto issue = [&](int buf, int j0){
    async_copy_b128(&sK[buf][eK],           &gK[(size_t)j0 * DHEAD + eK]);
    async_copy_b128(&sV[buf][dV * 32 + cV], &gV[(size_t)dV * SEQ + j0 + cV]);
  };

  issue(0, 0);
  for (int n = 0; n < SEQ / 32; ++n){
    const int cur = n & 1;
    if ((n + 1) * 32 < SEQ){
      issue(cur ^ 1, (n + 1) * 32);                      // prefetch next chunk
      asm volatile("s_wait_asynccnt 0x2" ::: "memory");  // current chunk landed
    } else {
      asm volatile("s_wait_asynccnt 0x0" ::: "memory");
    }
    __syncthreads();

    // S^T tiles: 2 j-tiles x (K=64 as 2x32)
    v8f st[2];
    for (int t = 0; t < 2; ++t){
      FragB a0, a1;
      int j = t * 16 + l15;
      a0.q[0] = *(const uint4*)&sK[cur][j * 64 + (hi ? 16 : 0)];
      a0.q[1] = *(const uint4*)&sK[cur][j * 64 + (hi ? 16 : 0) + 8];
      a1.q[0] = *(const uint4*)&sK[cur][j * 64 + 32 + (hi ? 16 : 0)];
      a1.q[1] = *(const uint4*)&sK[cur][j * 64 + 32 + (hi ? 16 : 0) + 8];
      v8f s = {};
      s = wmma_bf16(a0, q0, s);
      s = wmma_bf16(a1, q1, s);
      st[t] = s;
    }

    // online softmax (per query column i = lane&15; halves merged via shfl)
    float mloc = -3.0e38f;
    for (int t = 0; t < 2; ++t)
      for (int r = 0; r < 8; ++r) mloc = fmaxf(mloc, st[t][r]);
    mloc = fmaxf(mloc, __shfl_xor(mloc, 16, 32));
    float mnew = fmaxf(mrun, mloc);
    float corr = __expf(mrun - mnew);
    lrun *= corr;
    for (int c = 0; c < 4; ++c)
      for (int r = 0; r < 8; ++r) acc[c][r] *= corr;
    mrun = mnew;

    unsigned short* pw = sP[wave];
    for (int t = 0; t < 2; ++t){
      float p[8]; float ps = 0.f;
      for (int r = 0; r < 8; ++r){ p[r] = __expf(st[t][r] - mnew); ps += p[r]; }
      lrun += ps;
      uint4 pk;
      pk.x = (unsigned)f2bf(p[0]) | ((unsigned)f2bf(p[1]) << 16);
      pk.y = (unsigned)f2bf(p[2]) | ((unsigned)f2bf(p[3]) << 16);
      pk.z = (unsigned)f2bf(p[4]) | ((unsigned)f2bf(p[5]) << 16);
      pk.w = (unsigned)f2bf(p[6]) | ((unsigned)f2bf(p[7]) << 16);
      *(uint4*)&pw[l15 * 32 + t * 16 + (hi ? 8 : 0)] = pk;
    }

    FragB pb;   // P as B-fragment: row i = lane&15, 16 contiguous j
    pb.q[0] = *(const uint4*)&pw[l15 * 32 + (hi ? 16 : 0)];
    pb.q[1] = *(const uint4*)&pw[l15 * 32 + (hi ? 16 : 0) + 8];

    for (int c = 0; c < 4; ++c){                          // O^T += V x P^T
      FragB av;
      int d = c * 16 + l15;
      av.q[0] = *(const uint4*)&sV[cur][d * 32 + (hi ? 16 : 0)];
      av.q[1] = *(const uint4*)&sV[cur][d * 32 + (hi ? 16 : 0) + 8];
      acc[c] = wmma_bf16(av, pb, acc[c]);
    }
    __syncthreads();                                      // protect buf before reuse
  }

  float lsum = lrun + __shfl_xor(lrun, 16, 32);
  float rinv = 1.0f / lsum;
  int h = bh % HEADS;
  int b = bh / HEADS;
  for (int c = 0; c < 4; ++c){
    for (int r = 0; r < 8; ++r){
      int d = c * 16 + r + (hi ? 8 : 0);
      float v = acc[c][r] * rinv;
      int flat = i * DHEAD + d;                           // torch-faithful reshape
      int cc = h * DHEAD + (flat >> 11);
      int ll = flat & 2047;
      attnT[((size_t)b * SEQ + ll) * DIMC + cc] = f2bf(v);
    }
  }
}

// ---------------- output projection ----------------
// out[b][o][l] = sum_c wout[o][c]*attn[b][c][l] + bout[o]
__global__ void out_proj(const unsigned short* __restrict__ wob,
                         const unsigned short* __restrict__ attnT,
                         const float* __restrict__ bout,
                         float* __restrict__ out){
  const int lane = threadIdx.x & 31;
  const int wave = threadIdx.x >> 5;
  int task = blockIdx.x * (blockDim.x >> 5) + wave;       // B * 32 * 64 tasks
  int b  = task / (32 * 64);
  int rm = task % (32 * 64);
  int ot = (rm / 64) * 16;
  int lt = (rm % 64) * 32;
  const int hi  = (lane >= 16);
  const int l15 = lane & 15;

  v8f acc0 = {}, acc1 = {};
  int arow = ot + l15;
  for (int c = 0; c < DIMC; c += 32){
    FragB fa, fb0, fb1;
    fa.q[0] = *(const uint4*)&wob[arow * DIMC + c + (hi ? 16 : 0)];
    fa.q[1] = *(const uint4*)&wob[arow * DIMC + c + (hi ? 16 : 0) + 8];
    size_t r0 = ((size_t)b * SEQ + lt + l15) * DIMC + c + (hi ? 16 : 0);
    fb0.q[0] = *(const uint4*)&attnT[r0];
    fb0.q[1] = *(const uint4*)&attnT[r0 + 8];
    size_t r1 = ((size_t)b * SEQ + lt + 16 + l15) * DIMC + c + (hi ? 16 : 0);
    fb1.q[0] = *(const uint4*)&attnT[r1];
    fb1.q[1] = *(const uint4*)&attnT[r1 + 8];
    acc0 = wmma_bf16(fa, fb0, acc0);
    acc1 = wmma_bf16(fa, fb1, acc1);
  }
  for (int t = 0; t < 2; ++t){
    v8f acc = t ? acc1 : acc0;
    int l = lt + t * 16 + l15;
    for (int r = 0; r < 8; ++r){
      int o = ot + r + (hi ? 8 : 0);
      out[((size_t)b * DIMC + o) * SEQ + l] = acc[r] + bout[o];
    }
  }
}

// ---------------- launch ----------------
extern "C" void kernel_launch(void* const* d_in, const int* in_sizes, int n_in,
                              void* d_out, int out_size, void* d_ws, size_t ws_size,
                              hipStream_t stream) {
  const float* x     = (const float*)d_in[0];
  const float* w_qkv = (const float*)d_in[1];
  const float* w_out = (const float*)d_in[2];
  const float* b_out = (const float*)d_in[3];
  float* out = (float*)d_out;

  char* ws = (char*)d_ws;
  size_t off = 0;
  unsigned short* xT  = (unsigned short*)(ws + off); off += (size_t)BATCH*SEQ*DIMC*2;
  unsigned short* wb  = (unsigned short*)(ws + off); off += (size_t)3*HID*DIMC*2;
  unsigned short* wob = (unsigned short*)(ws + off); off += (size_t)DIMC*HID*2;
  unsigned short* Qs  = (unsigned short*)(ws + off); off += (size_t)BATCH*HEADS*SEQ*DHEAD*2;
  unsigned short* Kb  = (unsigned short*)(ws + off); off += (size_t)BATCH*HEADS*SEQ*DHEAD*2;
  unsigned short* Vb  = (unsigned short*)(ws + off); off += (size_t)BATCH*HEADS*SEQ*DHEAD*2;
  unsigned short* aT  = (unsigned short*)(ws + off); off += (size_t)BATCH*SEQ*DIMC*2;

  int n1 = 3 * HID * DIMC;
  cvt_f32_bf16<<<(n1 + 255) / 256, 256, 0, stream>>>(w_qkv, wb, n1);
  int n2 = DIMC * HID;
  cvt_f32_bf16<<<(n2 + 255) / 256, 256, 0, stream>>>(w_out, wob, n2);
  int n3 = BATCH * DIMC * SEQ;
  transpose_x<<<(n3 + 255) / 256, 256, 0, stream>>>(x, xT);

  qkv_gemm<<<(BATCH * 96 * 64) / 8, 256, 0, stream>>>(wb, xT, Qs, Kb, Vb);
  flash_attn<<<BATCH * HEADS * (SEQ / 128), 256, 0, stream>>>(Qs, Kb, Vb, aT);
  out_proj<<<(BATCH * 32 * 64) / 8, 256, 0, stream>>>(wob, aT, b_out, out);
}
